// OutliersQLinearColumn_20675972563027
// MI455X (gfx1250) — compile-verified, matrix operand order checked by reading, not run on hardware
//
#include <hip/hip_runtime.h>
#include <hip/hip_bf16.h>

// ---------------------------------------------------------------------------
// XNOR-binarized linear with outlier columns, for MI455X (gfx1250, wave32).
//   out = x @ wq.T + bias,  wq = sign(w - rowmean) * rowmean|.|, outlier cols
//   replaced by full-precision outlier_weight.
// wq is exactly (+/- scale) per row -> bf16 representation; GEMM runs as
// v_wmma_f32_16x16x32_bf16 with f32 accumulation. Tile feed uses the CDNA5
// async global->LDS path (global_load_async_to_lds_b128 + s_wait_asynccnt)
// so no VGPR staging is live across the WMMA region (no spills).
// ---------------------------------------------------------------------------

typedef __attribute__((ext_vector_type(16))) __bf16 v16bf;
typedef __attribute__((ext_vector_type(8)))  __bf16 v8bf;
typedef __attribute__((ext_vector_type(4)))  __bf16 v4bf;
typedef __attribute__((ext_vector_type(8)))  float  v8f;
typedef __attribute__((ext_vector_type(4)))  int    v4i;

// Pointer types the async-LDS builtin expects (from hipcc diagnostics):
//   global side: addrspace(1) int4*, LDS side: addrspace(3) int4*
typedef __attribute__((address_space(1))) v4i gv4i;
typedef __attribute__((address_space(3))) v4i lv4i;

#define BM 128
#define BN 128
#define BK 32
#define LDT 40   // padded LDS row stride (bf16): 80B rows, keeps 16B alignment

// ---------------------------------------------------------------------------
// Kernel 1: f32 -> bf16 convert (x matrix), 4 elems/thread, 128-bit loads.
// ---------------------------------------------------------------------------
__global__ void cvt_f32_bf16(const float* __restrict__ x,
                             __bf16* __restrict__ o, long n) {
  long i = ((long)blockIdx.x * blockDim.x + threadIdx.x) * 4;
  if (i + 3 < n) {
    float4 v = *(const float4*)(x + i);
    v4bf p = { (__bf16)v.x, (__bf16)v.y, (__bf16)v.z, (__bf16)v.w };
    *(v4bf*)(o + i) = p;
  }
}

// ---------------------------------------------------------------------------
// Kernel 2: per-output-row binarize: wc = w - mean(row); scale = mean|wc|;
// wq = sign(wc)*scale (sign(0)=0 like jnp.sign); then splice outlier columns.
// ---------------------------------------------------------------------------
__global__ __launch_bounds__(256)
void xnor_quantize_w(const float* __restrict__ w,
                     const float* __restrict__ ow,
                     const int*   __restrict__ idx,
                     __bf16* __restrict__ wb,
                     int K, int NOUT) {
  __shared__ float red[256];
  const int row = blockIdx.x;
  const int t = threadIdx.x;
  const float* wr = w + (size_t)row * K;

  float s = 0.f;
  for (int i = t; i < K; i += 256) s += wr[i];
  red[t] = s;
  __syncthreads();
  for (int off = 128; off > 0; off >>= 1) {
    if (t < off) red[t] += red[t + off];
    __syncthreads();
  }
  const float mean = red[0] / (float)K;
  __syncthreads();

  float a = 0.f;
  for (int i = t; i < K; i += 256) a += fabsf(wr[i] - mean);
  red[t] = a;
  __syncthreads();
  for (int off = 128; off > 0; off >>= 1) {
    if (t < off) red[t] += red[t + off];
    __syncthreads();
  }
  const float scale = red[0] / (float)K;

  __bf16* wbr = wb + (size_t)row * K;
  for (int i = t; i < K; i += 256) {
    float wc = wr[i] - mean;
    float q = (wc > 0.f) ? scale : ((wc < 0.f) ? -scale : 0.f);
    wbr[i] = (__bf16)q;
  }
  __syncthreads();  // dense writes must land before outlier splice
  for (int j = t; j < NOUT; j += 256)
    wbr[idx[j]] = (__bf16)ow[(size_t)row * NOUT + j];
}

// ---------------------------------------------------------------------------
// Kernel 3: C[T,N] = A[T,K](bf16) * Bw[N,K]^T(bf16) + bias, f32 out.
// 128x128 block tile, BK=32, double-buffered LDS via async global->LDS,
// 8 wave32s in a 4(M)x2(N) wave grid; each wave: 2x4 16x16 tiles = 8 WMMAs.
// ---------------------------------------------------------------------------
static __device__ __forceinline__ v16bf lds_load_frag(const __bf16* p) {
  // two 16B LDS reads (ds_load_b128) -> one 16-elem bf16 fragment
  const v8bf* q = (const v8bf*)p;
  v8bf lo = q[0];
  v8bf hi = q[1];
  return __builtin_shufflevector(lo, hi, 0, 1, 2, 3, 4, 5, 6, 7,
                                        8, 9, 10, 11, 12, 13, 14, 15);
}

__global__ __launch_bounds__(256)
void xnor_gemm_bf16_wmma(const __bf16* __restrict__ A,
                         const __bf16* __restrict__ Bw,
                         const float* __restrict__ bias,
                         float* __restrict__ C,
                         int T, int N, int K) {
  __shared__ __bf16 As[2][BM * LDT];
  __shared__ __bf16 Bs[2][BM * LDT];

  const int t    = threadIdx.x;
  const int lane = t & 31;
  const int wave = t >> 5;
  const int wm   = (wave & 3) * 32;   // wave M offset in block tile
  const int wn   = (wave >> 2) * 64;  // wave N offset in block tile
  const int fr   = lane & 15;         // fragment row within 16x16 tile
  const int fk   = (lane >> 4) << 4;  // K half selected by lane group

  const int m0 = blockIdx.x * BM;
  const int n0 = blockIdx.y * BN;

  // Tile feed: per thread, 2 x 16B chunks per matrix per K-step:
  //   chunk c: row = (t>>2) + c*64, col = (t&3)*8 bf16.
  const int rloc = t >> 2;
  const int cloc = (t & 3) * 8;

  // 4 async b128 issues per thread per K-step; no VGPR staging.
  auto aload = [&](int buf, int k0) {
#pragma unroll
    for (int c = 0; c < 2; ++c) {
      const int r = rloc + c * 64;
      __builtin_amdgcn_global_load_async_to_lds_b128(
          (gv4i*)(A + (size_t)(m0 + r) * K + k0 + cloc),
          (lv4i*)(&As[buf][r * LDT + cloc]), 0, 0);
      __builtin_amdgcn_global_load_async_to_lds_b128(
          (gv4i*)(Bw + (size_t)(n0 + r) * K + k0 + cloc),
          (lv4i*)(&Bs[buf][r * LDT + cloc]), 0, 0);
    }
  };

  v8f acc[2][4];
  const v8f vzero = {0.f, 0.f, 0.f, 0.f, 0.f, 0.f, 0.f, 0.f};
#pragma unroll
  for (int i = 0; i < 2; ++i)
#pragma unroll
    for (int j = 0; j < 4; ++j) acc[i][j] = vzero;

  aload(0, 0);  // prologue: buffer 0 in flight

  const int nk = K >> 5;
  for (int kb = 0; kb < nk; ++kb) {
    const int cur = kb & 1;
    if (kb + 1 < nk) {
      aload(cur ^ 1, (kb + 1) << 5);       // next tile in flight
      __builtin_amdgcn_s_wait_asynccnt(4); // in-order: buf-cur's 4 are done
    } else {
      __builtin_amdgcn_s_wait_asynccnt(0);
    }
    if (kb + 2 < nk) {  // L2 prefetch two K-steps ahead (global_prefetch_b8)
      const int kp = (kb + 2) << 5;
      __builtin_prefetch(A  + (size_t)(m0 + rloc) * K + kp + cloc, 0, 3);
      __builtin_prefetch(Bw + (size_t)(n0 + rloc) * K + kp + cloc, 0, 3);
    }
    __syncthreads();  // buf cur visible to all waves

    v16bf av[2], bv[4];
#pragma unroll
    for (int tm = 0; tm < 2; ++tm)
      av[tm] = lds_load_frag(&As[cur][(wm + tm * 16 + fr) * LDT + fk]);
#pragma unroll
    for (int tn = 0; tn < 4; ++tn)
      bv[tn] = lds_load_frag(&Bs[cur][(wn + tn * 16 + fr) * LDT + fk]);

#pragma unroll
    for (int tm = 0; tm < 2; ++tm)
#pragma unroll
      for (int tn = 0; tn < 4; ++tn)
        acc[tm][tn] = __builtin_amdgcn_wmma_f32_16x16x32_bf16(
            false, av[tm], false, bv[tn], (short)0, acc[tm][tn], false, false);

    __syncthreads();  // reads done before buf cur gets overwritten next iter
  }

  // Epilogue: 16x16 f32 C layout is 8 VGPRs; lanes 0-15 -> M=r, 16-31 -> M=8+r.
  const int rbase = (lane >> 4) * 8;
  const int cn    = lane & 15;
#pragma unroll
  for (int tm = 0; tm < 2; ++tm) {
#pragma unroll
    for (int tn = 0; tn < 4; ++tn) {
      const int col = n0 + wn + tn * 16 + cn;
      const float b = bias[col];
#pragma unroll
      for (int r = 0; r < 8; ++r) {
        const int row = m0 + wm + tm * 16 + rbase + r;
        C[(size_t)row * N + col] = acc[tm][tn][r] + b;
      }
    }
  }
}

// ---------------------------------------------------------------------------
// Host launcher
// ---------------------------------------------------------------------------
extern "C" void kernel_launch(void* const* d_in, const int* in_sizes, int n_in,
                              void* d_out, int out_size, void* d_ws, size_t ws_size,
                              hipStream_t stream) {
  const float* x    = (const float*)d_in[0];  // [T, IC]
  const float* w    = (const float*)d_in[1];  // [OC, IC]
  const float* bias = (const float*)d_in[2];  // [OC]
  const float* ow   = (const float*)d_in[3];  // [OC, NOUT]
  const int*   idx  = (const int*)d_in[4];    // [NOUT]

  const int OC   = in_sizes[2];
  const int NOUT = in_sizes[4];
  const int IC   = in_sizes[1] / OC;
  const int T    = in_sizes[0] / IC;

  // Workspace: bf16 x [T*IC] then bf16 wq [OC*IC] (64 MB total at 4096^2).
  __bf16* xb = (__bf16*)d_ws;
  __bf16* wb = xb + (size_t)T * IC;

  {
    long n = (long)T * IC;
    long nthreads = n / 4;
    int blocks = (int)((nthreads + 255) / 256);
    cvt_f32_bf16<<<blocks, 256, 0, stream>>>(x, xb, n);
  }
  xnor_quantize_w<<<OC, 256, 0, stream>>>(w, ow, idx, wb, IC, NOUT);

  dim3 grid(T / BM, OC / BN);
  xnor_gemm_bf16_wmma<<<grid, 256, 0, stream>>>(xb, wb, bias, (float*)d_out,
                                                T, OC, IC);
}
